// BigBirdTransformerBlock_34497177321774
// MI455X (gfx1250) — matrix operand order repaired and checked by reading, not compiled, not run on hardware
//
#include <hip/hip_runtime.h>
#include <hip/hip_bf16.h>

typedef __attribute__((ext_vector_type(16))) _Float16 v16h;
typedef __attribute__((ext_vector_type(8)))  float    v8f;
typedef int v4i __attribute__((vector_size(16)));     // GCC-style, matches builtin

#define DEV __device__ __forceinline__

constexpr int   NMID = 60;          // S/BLK - g/BLK - 2 = 64-2-2
constexpr float RS   = 0.125f;      // 1/sqrt(64)
constexpr float PEN  = -10000.0f;

// Padded LDS row strides (in halfs) -> 16B aligned, bank-conflict-free
constexpr int KVS = 72;   // 144B rows for 64-wide K/V/P tiles
constexpr int BS  = 40;   // 80B rows for 32-wide B tiles

#if __has_builtin(__builtin_amdgcn_global_load_async_to_lds_b128)
#define HAVE_ASYNC 1
#endif

DEV void async_copy_b128(const _Float16* g, _Float16* l) {
#ifdef HAVE_ASYNC
  __attribute__((address_space(1))) void* gp =
      (__attribute__((address_space(1))) void*)(const_cast<_Float16*>(g));
  __attribute__((address_space(3))) void* lp =
      (__attribute__((address_space(3))) void*)(l);
  __builtin_amdgcn_global_load_async_to_lds_b128(
      (__attribute__((address_space(1))) v4i*)gp,
      (__attribute__((address_space(3))) v4i*)lp, 0, 0);
#else
  *(int4*)l = *(const int4*)g;
#endif
}

DEV void async_wait() {
#ifdef HAVE_ASYNC
#if __has_builtin(__builtin_amdgcn_s_wait_asynccnt)
  __builtin_amdgcn_s_wait_asynccnt(0);
#else
  asm volatile("s_wait_asynccnt 0" ::: "memory");
#endif
#endif
}

DEV int laneid() { return threadIdx.x & 31; }

DEV v8f zero8() { v8f z;
#pragma unroll
  for (int i = 0; i < 8; ++i) z[i] = 0.0f;
  return z; }

// A-operand (16x32 f16): lane = row M (mod 16); half-wave selects K-groups:
// lanes 0-15 hold K 0..7 / 16..23, lanes 16-31 hold K 8..15 / 24..31.
DEV v16h load_a(const _Float16* p, int stride) {
  const int l = laneid();
  const _Float16* r = p + (size_t)(l & 15) * stride + (l >> 4) * 8;
  v16h o;
#pragma unroll
  for (int j = 0; j < 8; ++j) { o[j] = r[j]; o[8 + j] = r[16 + j]; }
  return o;
}

// B-operand (32x16 f16) from B^T stored row-major [n][k]:
// lane = column N (mod 16); lanes 0-15 hold K 0..7 / 16..23, lanes 16-31 hold K 8..15 / 24..31.
DEV v16h load_b(const _Float16* p, int stride) {
  const int l = laneid();
  const _Float16* r = p + (size_t)(l & 15) * stride + (l >> 4) * 8;
  v16h o;
#pragma unroll
  for (int j = 0; j < 8; ++j) { o[j] = r[j]; o[8 + j] = r[16 + j]; }
  return o;
}

DEV v8f wmma_f16(v16h a, v16h b, v8f c) {
  return __builtin_amdgcn_wmma_f32_16x16x32_f16(false, a, false, b, (short)0, c,
                                                false, false);
}

// ---------------------------------------------------------------- converts
__global__ void cvt_tokens(const float* __restrict__ x, _Float16* __restrict__ y, int n) {
  int i = blockIdx.x * 256 + threadIdx.x;
  if (i < n) y[i] = (_Float16)x[i];
}

// W[k][n] (f32, 1024x1024) -> Wt[n][k] (f16)
__global__ void cvt_wt(const float* __restrict__ w, _Float16* __restrict__ wt) {
  int i = blockIdx.x * 256 + threadIdx.x;
  int k = i >> 10, n = i & 1023;
  wt[(size_t)n * 1024 + k] = (_Float16)w[i];
}

// ---------------------------------------------------------------- GEMM
// C[8192][1024] = A[8192][1024] * Bt^T.  Block = 4 waves sharing one 64-col
// N-strip; each wave owns a 64x64 C tile.  B tiles (64x32) staged to LDS with
// async copies, double-buffered across the K loop.
template <int MODE>
__global__ void __launch_bounds__(128) gemm64(const _Float16* __restrict__ A,
                                              const _Float16* __restrict__ Bt,
                                              void* __restrict__ out,
                                              const float* __restrict__ bias) {
  __shared__ __align__(16) _Float16 bls[2][64 * BS];

  const int t    = threadIdx.x;
  const int widx = t >> 5, l = t & 31;
  const int n0   = (blockIdx.x & 15) * 64;           // shared by all 4 waves
  const int m0   = ((blockIdx.x >> 4) * 4 + widx) * 64;

  auto stage_b = [&](int k0, _Float16* buf) {
#pragma unroll
    for (int i = 0; i < 2; ++i) {
      int slot = t + i * 128;                        // 256 slots of 16B
      int row = slot >> 2, col8 = (slot & 3) * 8;
      async_copy_b128(Bt + (size_t)(n0 + row) * 1024 + k0 + col8,
                      buf + row * BS + col8);
    }
  };

  v8f C[4][4];
#pragma unroll
  for (int i = 0; i < 4; ++i)
#pragma unroll
    for (int j = 0; j < 4; ++j) C[i][j] = zero8();

  stage_b(0, &bls[0][0]);
  for (int kk = 0; kk < 32; ++kk) {
    const int k0 = kk * 32;
    async_wait();
    __syncthreads();
    if (kk + 1 < 32) stage_b(k0 + 32, &bls[(kk + 1) & 1][0]);

    v16h a[4];
#pragma unroll
    for (int mt = 0; mt < 4; ++mt)
      a[mt] = load_a(A + (size_t)(m0 + mt * 16) * 1024 + k0, 1024);

    const _Float16* bb = &bls[kk & 1][0];
#pragma unroll
    for (int nt = 0; nt < 4; ++nt) {
      v16h bf = load_b(bb + nt * 16 * BS, BS);
#pragma unroll
      for (int mt = 0; mt < 4; ++mt) C[mt][nt] = wmma_f16(a[mt], bf, C[mt][nt]);
    }
  }

#pragma unroll
  for (int mt = 0; mt < 4; ++mt)
#pragma unroll
    for (int nt = 0; nt < 4; ++nt)
#pragma unroll
      for (int r = 0; r < 8; ++r) {
        int   m = m0 + mt * 16 + r + ((l >> 4) << 3);
        int   n = n0 + nt * 16 + (l & 15);
        float v = C[mt][nt][r];
        if constexpr (MODE == 0) {
          int b = m >> 12, s = m & 4095, h = n >> 6, d = n & 63;
          ((_Float16*)out)[((size_t)((b * 16 + h) * 4096 + s)) * 64 + d] = (_Float16)v;
        } else if constexpr (MODE == 1) {
          int b = m >> 12, s = m & 4095, h = n >> 6, d = n & 63;
          ((_Float16*)out)[((size_t)((b * 16 + h) * 64 + d)) * 4096 + s] = (_Float16)v;
        } else {
          ((float*)out)[(size_t)m * 1024 + n] = v + bias[n];
        }
      }
}

// ---------------------------------------------------------------- attention
// grid (64 qblocks, 16 heads, 2 batch), 128 threads = 4 waves; wave = 16 rows.
// K/V 64-key chunks async-staged to LDS, double-buffered; flash-style softmax.
__global__ void __launch_bounds__(128) bigbird_attn(
    const _Float16* __restrict__ Q, const _Float16* __restrict__ K,
    const _Float16* __restrict__ Vt, const float* __restrict__ to_mask,
    const float* __restrict__ band_mask, const float* __restrict__ from_mask,
    _Float16* __restrict__ ctx) {
  __shared__ __align__(16) _Float16 kls[2][64 * KVS];
  __shared__ __align__(16) _Float16 vls[2][64 * KVS];
  __shared__ __align__(16) _Float16 plds[4][16 * KVS];

  const int qb = blockIdx.x, h = blockIdx.y, b = blockIdx.z;
  const int t = threadIdx.x, w = t >> 5, l = t & 31;
  const size_t hb = (size_t)(b * 16 + h);
  const _Float16* Qh = Q + hb * 4096 * 64;
  const _Float16* Kh = K + hb * 4096 * 64;
  const _Float16* Vh = Vt + hb * 64 * 4096;
  const int s0 = qb * 64 + w * 16;

  auto chunk_base = [&](int ci, int* bmode) -> int {
    *bmode = 0;
    if (qb > 2 && qb < 63 && ci >= 2) { *bmode = 1; return (qb - 3 + ci) * 64; }
    if (qb == 63 && ci >= 2) return (59 + ci) * 64;
    return ci * 64;
  };

  auto stage_chunk = [&](int base, _Float16* kbuf, _Float16* vbuf) {
#pragma unroll
    for (int i = 0; i < 4; ++i) {
      int slot = t + i * 128;                        // 512 slots of 16B
      int row = slot >> 3, col8 = (slot & 7) * 8;
      async_copy_b128(Kh + (size_t)(base + row) * 64 + col8,
                      kbuf + row * KVS + col8);
      async_copy_b128(Vh + (size_t)row * 4096 + base + col8,
                      vbuf + row * KVS + col8);
    }
  };

  v16h qa[2];
#pragma unroll
  for (int ks = 0; ks < 2; ++ks) qa[ks] = load_a(Qh + (size_t)s0 * 64 + ks * 32, 64);

  v8f   O[4];
  float rmax[8], rsum[8];
#pragma unroll
  for (int i = 0; i < 4; ++i) O[i] = zero8();
#pragma unroll
  for (int r = 0; r < 8; ++r) { rmax[r] = -1e30f; rsum[r] = 0.0f; }

  const int nch = (qb < 2) ? 64 : 5;
  int bm0;
  stage_chunk(chunk_base(0, &bm0), &kls[0][0], &vls[0][0]);

  for (int ci = 0; ci < nch; ++ci) {
    int bmode;
    const int base = chunk_base(ci, &bmode);

    async_wait();
    __syncthreads();
    if (ci + 1 < nch) {
      int bm;
      stage_chunk(chunk_base(ci + 1, &bm), &kls[(ci + 1) & 1][0],
                  &vls[(ci + 1) & 1][0]);
    }
    const _Float16* kb = &kls[ci & 1][0];
    const _Float16* vb = &vls[ci & 1][0];

    // ---- scores: S = Q * Kchunk^T
    v8f S[4];
#pragma unroll
    for (int nt = 0; nt < 4; ++nt) {
      v8f acc = zero8();
#pragma unroll
      for (int ks = 0; ks < 2; ++ks) {
        v16h bf = load_b(kb + (nt * 16) * KVS + ks * 32, KVS);
        acc = wmma_f16(qa[ks], bf, acc);
      }
      S[nt] = acc;
    }

    // ---- scale + additive mask
#pragma unroll
    for (int nt = 0; nt < 4; ++nt) {
      const int key = base + nt * 16 + (l & 15);
      float     mb  = 0.0f;
      if (!bmode) mb = (1.0f - to_mask[b * 4096 + key]) * PEN;
#pragma unroll
      for (int r = 0; r < 8; ++r) {
        float x = S[nt][r] * RS;
        if (bmode) {
          int qloc = w * 16 + r + ((l >> 4) << 3);
          int mi   = qb - 3;
          int kk   = key - (qb - 1) * 64;   // position inside 3*BLK band
          x += (1.0f - band_mask[((size_t)(b * NMID + mi) * 64 + qloc) * 192 + kk]) * PEN;
        } else {
          x += mb;
        }
        S[nt][r] = x;
      }
    }

    // ---- online softmax (16-lane reductions; row = half-wave lane group)
    float cmax[8], corr[8], csum[8];
#pragma unroll
    for (int r = 0; r < 8; ++r) {
      float m = S[0][r];
#pragma unroll
      for (int nt = 1; nt < 4; ++nt) m = fmaxf(m, S[nt][r]);
#pragma unroll
      for (int off = 1; off < 16; off <<= 1) m = fmaxf(m, __shfl_xor(m, off, 16));
      cmax[r] = m;
    }
#pragma unroll
    for (int r = 0; r < 8; ++r) {
      float nm = fmaxf(rmax[r], cmax[r]);
      corr[r]  = __expf(rmax[r] - nm);
      rmax[r]  = nm;
      csum[r]  = 0.0f;
    }
#pragma unroll
    for (int nt = 0; nt < 4; ++nt)
#pragma unroll
      for (int r = 0; r < 8; ++r) {
        float p = __expf(S[nt][r] - rmax[r]);
        S[nt][r] = p;
        csum[r] += p;
      }
#pragma unroll
    for (int r = 0; r < 8; ++r) {
      float cs = csum[r];
#pragma unroll
      for (int off = 1; off < 16; off <<= 1) cs += __shfl_xor(cs, off, 16);
      rsum[r] = rsum[r] * corr[r] + cs;
    }
#pragma unroll
    for (int nt = 0; nt < 4; ++nt)
#pragma unroll
      for (int r = 0; r < 8; ++r) O[nt][r] *= corr[r];

    // ---- stage P through LDS (C-layout -> row-major) for A-operand reload
#pragma unroll
    for (int nt = 0; nt < 4; ++nt)
#pragma unroll
      for (int r = 0; r < 8; ++r) {
        int row = r + ((l >> 4) << 3);
        int col = nt * 16 + (l & 15);
        plds[w][row * KVS + col] = (_Float16)S[nt][r];
      }

    // ---- O += P * Vchunk (V staged as [d][key])
#pragma unroll
    for (int ks = 0; ks < 2; ++ks) {
      v16h pa = load_a(&plds[w][ks * 32], KVS);
#pragma unroll
      for (int nt = 0; nt < 4; ++nt) {
        v16h bf = load_b(vb + (nt * 16) * KVS + ks * 32, KVS);
        O[nt] = wmma_f16(pa, bf, O[nt]);
      }
    }
  }

  // ---- normalize, from_mask, store ctx (f16, [b,s, h*64+d])
#pragma unroll
  for (int r = 0; r < 8; ++r) {
    int   s   = s0 + r + ((l >> 4) << 3);
    float fm  = from_mask[b * 4096 + s];
    float inv = fm / rsum[r];
#pragma unroll
    for (int nt = 0; nt < 4; ++nt) {
      int d = nt * 16 + (l & 15);
      ctx[((size_t)(b * 4096 + s)) * 1024 + h * 64 + d] = (_Float16)(O[nt][r] * inv);
    }
  }
}

// ---------------------------------------------------------------- launch
extern "C" void kernel_launch(void* const* d_in, const int* in_sizes, int n_in,
                              void* d_out, int out_size, void* d_ws, size_t ws_size,
                              hipStream_t stream) {
  const float* tokens    = (const float*)d_in[0];
  const float* band_mask = (const float*)d_in[1];
  const float* from_mask = (const float*)d_in[2];
  const float* to_mask   = (const float*)d_in[3];
  const float* Wq        = (const float*)d_in[4];
  const float* Wk        = (const float*)d_in[5];
  const float* Wv        = (const float*)d_in[6];
  const float* Wu        = (const float*)d_in[7];
  const float* bu        = (const float*)d_in[8];
  float*       out       = (float*)d_out;

  char*  ws  = (char*)d_ws;
  size_t off = 0;
  auto alloc = [&](size_t bytes) -> void* {
    void* p = ws + off;
    off += (bytes + 255) & ~(size_t)255;
    return p;
  };

  const size_t tokelems = 8192ull * 1024;       // B*S x D
  _Float16* tok16 = (_Float16*)alloc(tokelems * 2);
  _Float16* wqt   = (_Float16*)alloc(1024ull * 1024 * 2);
  _Float16* wkt   = (_Float16*)alloc(1024ull * 1024 * 2);
  _Float16* wvt   = (_Float16*)alloc(1024ull * 1024 * 2);
  _Float16* wut   = (_Float16*)alloc(1024ull * 1024 * 2);
  _Float16* q16   = (_Float16*)alloc(tokelems * 2);  // [b,h,s,d]
  _Float16* k16   = (_Float16*)alloc(tokelems * 2);  // [b,h,s,d]
  _Float16* vt16  = (_Float16*)alloc(tokelems * 2);  // [b,h,d,s]
  _Float16* ctx16 = (_Float16*)alloc(tokelems * 2);  // [b,s,h*d]

  cvt_tokens<<<(int)(tokelems / 256), 256, 0, stream>>>(tokens, tok16, (int)tokelems);
  cvt_wt<<<4096, 256, 0, stream>>>(Wq, wqt);
  cvt_wt<<<4096, 256, 0, stream>>>(Wk, wkt);
  cvt_wt<<<4096, 256, 0, stream>>>(Wv, wvt);
  cvt_wt<<<4096, 256, 0, stream>>>(Wu, wut);

  gemm64<0><<<512, 128, 0, stream>>>(tok16, wqt, (void*)q16, nullptr);
  gemm64<0><<<512, 128, 0, stream>>>(tok16, wkt, (void*)k16, nullptr);
  gemm64<1><<<512, 128, 0, stream>>>(tok16, wvt, (void*)vt16, nullptr);

  bigbird_attn<<<dim3(64, 16, 2), 128, 0, stream>>>(q16, k16, vt16, to_mask,
                                                    band_mask, from_mask, ctx16);

  gemm64<2><<<512, 128, 0, stream>>>(ctx16, wut, (void*)out, bu);
}
